// trajectory2seq_3058016715235
// MI455X (gfx1250) — matrix operand order repaired and checked by reading, not compiled, not run on hardware
//
#include <hip/hip_runtime.h>
#include <hip/hip_bf16.h>

// ---------------------------------------------------------------------------
// Trajectory2Seq on MI455X (gfx1250): bf16 WMMA everywhere a GEMM exists,
// LDS-resident RNN state in the encoder (ds ops), async Global->LDS staging
// of the per-step input slice (ASYNCcnt path), persistent-grid decoder with
// grid barrier (+ cluster-barrier fast path), wave32 shuffles.
// ---------------------------------------------------------------------------

typedef __bf16 bf16;
typedef __attribute__((ext_vector_type(16))) __bf16 v16bf;
typedef __attribute__((ext_vector_type(8)))  __bf16 v8bf;
typedef __attribute__((ext_vector_type(8)))  float  v8f;

#define WMMA_BF16(a, b, c) \
  __builtin_amdgcn_wmma_f32_16x16x32_bf16(false, (a), false, (b), (short)0, (c), false, false)

static constexpr int Bn = 128, Sn = 256, Hn = 512, Vn = 512, Tn = 64;
static constexpr int DEC_WG = 64, DEC_THR = 256;  // 512 waves total

// ---------------- small helpers ----------------
__device__ __forceinline__ bf16 f2bf(float f) {
  unsigned u = __builtin_bit_cast(unsigned, f);
  unsigned r = (u + 0x7FFFu + ((u >> 16) & 1u)) >> 16;
  return __builtin_bit_cast(bf16, (unsigned short)r);
}
__device__ __forceinline__ float bf2f(bf16 b) {
  unsigned u = ((unsigned)__builtin_bit_cast(unsigned short, b)) << 16;
  return __builtin_bit_cast(float, u);
}

// A-fragment (16x32 bf16) from row-major [ld] storage; Abase = tile row 0.
// lane&15 = row, lane>>4 selects K-half: elems 0-7 -> kb..kb+7, 8-15 -> kb+16..kb+23
__device__ __forceinline__ v16bf load_afrag(const bf16* Abase, int ld, int kk, int lane) {
  int row  = lane & 15;
  int kb   = kk * 32 + (lane >> 4) * 8;
  const bf16* p = Abase + (size_t)row * ld + kb;
  union { v16bf v; v8bf h[2]; } u;
  u.h[0] = *(const v8bf*)p;
  u.h[1] = *(const v8bf*)(p + 16);
  return u.v;
}

// B-fragment (32x16 bf16) from prepacked layout: per (nTile,kk), 32 lanes x 16 bf16.
__device__ __forceinline__ v16bf load_bfrag(const bf16* Bpk, int KC, int nTile, int kk, int lane) {
  const bf16* p = Bpk + (((size_t)nTile * KC + kk) * 32 + lane) * 16;
  return *(const v16bf*)p;
}

__device__ __forceinline__ float wave_max(float v) {
  for (int m = 16; m > 0; m >>= 1) v = fmaxf(v, __shfl_xor(v, m, 32));
  return v;
}
__device__ __forceinline__ float wave_sum(float v) {
  for (int m = 16; m > 0; m >>= 1) v += __shfl_xor(v, m, 32);
  return v;
}

// Grid barrier (sense reversal). Cluster-barrier fast path emits the CDNA5
// s_barrier_signal/-wait -3 pair; it is hardware-NOP when not cluster-dispatched.
__device__ __forceinline__ void grid_barrier(unsigned* cnt, unsigned* gen, unsigned nwg) {
  __syncthreads();
  if (threadIdx.x == 0) {
    __threadfence();
    volatile unsigned* vgen = (volatile unsigned*)gen;
    unsigned g = *vgen;
    __threadfence();
    unsigned prev = atomicAdd(cnt, 1u);
    if (prev == nwg - 1u) {
      atomicExch(cnt, 0u);
      __threadfence();
      atomicAdd(gen, 1u);
    } else {
      while (*vgen == g) { __builtin_amdgcn_s_sleep(2); }
    }
    __threadfence();
  }
  if (__builtin_amdgcn_cluster_id_x() != 0) {
    __builtin_amdgcn_s_cluster_barrier();
  }
  __syncthreads();
}

// ---------------- weight packing ----------------
// src: row-major (N,K) fp32, used as Bmat[k][n] = src[n][k].
// dst element i = (((nTile*KC + kk)*32 + lane)*16 + e), value src[n*K + k],
//   n = nTile*16 + (lane&15),  k = kk*32 + (e>>3)*16 + (lane>>4)*8 + (e&7).
__global__ void pack_b_kernel(const float* __restrict__ src, bf16* __restrict__ dst,
                              int N, int K) {
  int KC = K >> 5;
  int total = N * K;
  for (int i = blockIdx.x * blockDim.x + threadIdx.x; i < total;
       i += gridDim.x * blockDim.x) {
    int e = i & 15;
    int lane = (i >> 4) & 31;
    int t2 = i >> 9;
    int kk = t2 % KC;
    int nTile = t2 / KC;
    int n = nTile * 16 + (lane & 15);
    int k = kk * 32 + (e >> 3) * 16 + (lane >> 4) * 8 + (e & 7);
    dst[i] = f2bf(src[(size_t)n * K + k]);
  }
}

__global__ void f2bf_kernel(const float* __restrict__ src, bf16* __restrict__ dst, int n) {
  for (int i = blockIdx.x * blockDim.x + threadIdx.x; i < n; i += gridDim.x * blockDim.x)
    dst[i] = f2bf(src[i]);
}

// values = valF + valB  (in place into valF)
__global__ void addval_kernel(bf16* __restrict__ valF, const bf16* __restrict__ valB, int n) {
  for (int i = blockIdx.x * blockDim.x + threadIdx.x; i < n; i += gridDim.x * blockDim.x)
    valF[i] = f2bf(bf2f(valF[i]) + bf2f(valB[i]));
}

// ---------------- encoder: bidirectional tanh RNN ----------------
// gridDim.x = 2 (direction), blockDim = 1024 (32 waves on one WGP).
// Hidden state ping-pong in LDS (2 x 128KB bf16); the per-step x slice is
// DMA'd Global->LDS with global_load_async_to_lds_b32, double-buffered so the
// next step's slice streams while this step's WMMA GEMM runs.
__global__ __launch_bounds__(1024) void enc_rnn_kernel(
    const float* __restrict__ x,
    const float* __restrict__ Wih_f, const float* __restrict__ bias_f,
    const float* __restrict__ Wih_b, const float* __restrict__ bias_b,
    const bf16* __restrict__ pWhh_f, const bf16* __restrict__ pWhh_b,
    bf16* __restrict__ valF, bf16* __restrict__ valB,
    bf16* __restrict__ hfb) {
  extern __shared__ char smem_raw[];
  bf16* h0buf = (bf16*)smem_raw;
  bf16* h1buf = h0buf + Bn * Hn;
  float* xstage = (float*)(smem_raw + (size_t)2 * Bn * Hn * sizeof(bf16));  // 2 x 256 floats

  const int dir = blockIdx.x;
  const float* Wih  = dir ? Wih_b  : Wih_f;
  const float* bias = dir ? bias_b : bias_f;
  const bf16*  pWhh = dir ? pWhh_b : pWhh_f;
  bf16* valOut = dir ? valB : valF;

  // async-stage x[:, :, t(stp)] into LDS buffer (stp&1): one lane per element
  auto stage_x = [&](int stp_idx) {
    if (stp_idx < Sn && threadIdx.x < 2 * Bn) {
      int tcol = dir ? (Sn - 1 - stp_idx) : stp_idx;
      int b = threadIdx.x >> 1, c = threadIdx.x & 1;
      const float* gp = x + (size_t)b * 2 * Sn + c * Sn + tcol;
      float* lp = xstage + (stp_idx & 1) * (2 * Bn) + threadIdx.x;
      unsigned ldso = (unsigned)(unsigned long long)(size_t)lp;
      unsigned long long ga = (unsigned long long)(size_t)gp;
      asm volatile("global_load_async_to_lds_b32 %0, %1, off"
                   :: "v"(ldso), "v"(ga) : "memory");
    }
  };

  for (int i = threadIdx.x; i < Bn * Hn; i += blockDim.x) h0buf[i] = f2bf(0.f);
  stage_x(0);
  __syncthreads();

  bf16* hcur = h0buf;
  bf16* hnxt = h1buf;

  const int w = threadIdx.x >> 5, lane = threadIdx.x & 31;
  const int mTile = w & 7, ntBase = (w >> 3) * 8;  // 8 mTiles x 4 nt-groups
  const int laneN = lane & 15, half = lane >> 4;

  for (int stp = 0; stp < Sn; ++stp) {
    const int t = dir ? (Sn - 1 - stp) : stp;

    // wait for this step's async x slice, make it visible block-wide
    asm volatile("s_wait_asynccnt 0x0" ::: "memory");
    __syncthreads();

    const float* xb = xstage + (stp & 1) * (2 * Bn);
    float x0[8], x1[8];
#pragma unroll
    for (int r = 0; r < 8; ++r) {
      int b_r = mTile * 16 + r + 8 * half;
      x0[r] = xb[b_r * 2 + 0];   // ds_load_b32
      x1[r] = xb[b_r * 2 + 1];
    }
    stage_x(stp + 1);  // overlap next slice DMA with this step's GEMM

    for (int nt = 0; nt < 8; ++nt) {
      const int nG = ntBase + nt;
      const int n  = nG * 16 + laneN;
      const float wi0 = Wih[n * 2 + 0], wi1 = Wih[n * 2 + 1], bb = bias[n];
      v8f acc;
#pragma unroll
      for (int r = 0; r < 8; ++r) acc[r] = bb + x0[r] * wi0 + x1[r] * wi1;

      const int arow = mTile * 16 + laneN;
#pragma unroll
      for (int kk = 0; kk < Hn / 32; ++kk) {
        const int kb = kk * 32 + half * 8;
        union { v16bf v; v8bf h[2]; } u;
        u.h[0] = *(const v8bf*)&hcur[(size_t)arow * Hn + kb];        // ds_load_b128
        u.h[1] = *(const v8bf*)&hcur[(size_t)arow * Hn + kb + 16];   // ds_load_b128
        if (kk + 1 < Hn / 32)
          __builtin_prefetch(pWhh + (((size_t)nG * (Hn / 32) + kk + 1) * 32 + lane) * 16, 0, 1);
        v16bf bfr = load_bfrag(pWhh, Hn / 32, nG, kk, lane);
        acc = WMMA_BF16(u.v, bfr, acc);
      }
#pragma unroll
      for (int r = 0; r < 8; ++r) {
        int b_r = mTile * 16 + r + 8 * half;
        float hv = tanhf(acc[r]);
        bf16 hb = f2bf(hv);
        hnxt[(size_t)b_r * Hn + n] = hb;                             // ds_store_b16
        valOut[((size_t)b_r * Sn + t) * Hn + n] = hb;                // encoder output
      }
    }
    __syncthreads();
    bf16* tmp = hcur; hcur = hnxt; hnxt = tmp;
  }
  // final hidden state -> hfb[b][dir*H + h]  (bf16, concat layout for We2d GEMM)
  for (int i = threadIdx.x; i < Bn * Hn; i += blockDim.x) {
    int b = i >> 9, hh = i & (Hn - 1);
    hfb[(size_t)b * (2 * Hn) + dir * Hn + hh] = hcur[i];
  }
}

// ---------------- decoder init: hdec0 = relu([hf,hb]@We2d^T + be2d) ----------
__global__ __launch_bounds__(256) void dec_init_kernel(
    const bf16* __restrict__ hfb, const bf16* __restrict__ pWe2d,
    const float* __restrict__ be2d,
    bf16* __restrict__ hprev, int* __restrict__ tok, unsigned* __restrict__ bar) {
  const int lane = threadIdx.x & 31;
  const int gw = blockIdx.x * (blockDim.x >> 5) + (threadIdx.x >> 5);
  if (gw < 256) {
    const int mTile = gw >> 5, nTile = gw & 31;
    const int laneN = lane & 15, half = lane >> 4;
    const int n = nTile * 16 + laneN;
    v8f acc;
    const float bb = be2d[n];
#pragma unroll
    for (int r = 0; r < 8; ++r) acc[r] = bb;
    const bf16* Abase = hfb + (size_t)mTile * 16 * (2 * Hn);
#pragma unroll 4
    for (int kk = 0; kk < (2 * Hn) / 32; ++kk) {
      v16bf a   = load_afrag(Abase, 2 * Hn, kk, lane);
      v16bf bfr = load_bfrag(pWe2d, (2 * Hn) / 32, nTile, kk, lane);
      acc = WMMA_BF16(a, bfr, acc);
    }
#pragma unroll
    for (int r = 0; r < 8; ++r) {
      int b_r = mTile * 16 + r + 8 * half;
      hprev[(size_t)b_r * Hn + n] = f2bf(fmaxf(acc[r], 0.f));
    }
  }
  if (blockIdx.x == 0) {
    if (threadIdx.x < Bn) tok[threadIdx.x] = 0;
    if (threadIdx.x == 0) { bar[0] = 0u; bar[1] = 0u; }
  }
}

// ---------------- decoder: T persistent steps with grid barriers -------------
__global__ __launch_bounds__(DEC_THR) void dec_loop_kernel(
    const bf16* __restrict__ embbf,
    const bf16* __restrict__ pWih_d, const bf16* __restrict__ pWhh_d,
    const float* __restrict__ b_d,
    const bf16* __restrict__ pWq, const float* __restrict__ bq,
    const bf16* __restrict__ pWfc, const float* __restrict__ bfc,
    const bf16* __restrict__ values,
    bf16* hprevG, bf16* hnextG, bf16* __restrict__ hc,
    float* __restrict__ qbuf, float* __restrict__ scores, float* __restrict__ wbuf,
    int* __restrict__ tok,
    float* __restrict__ out_vec, float* __restrict__ out_hT, float* __restrict__ out_att,
    unsigned* bar) {
  const int lane = threadIdx.x & 31;
  const int gw = blockIdx.x * (blockDim.x >> 5) + (threadIdx.x >> 5);
  const int tid = blockIdx.x * blockDim.x + threadIdx.x;
  const int nthr = gridDim.x * blockDim.x;
  unsigned* cnt = bar;
  unsigned* gen = bar + 1;

  bf16* hprev = hprevG;
  bf16* hnext = hnextG;

  const int laneN = lane & 15, half = lane >> 4;

  for (int st = 0; st < Tn; ++st) {
    // -- phase A: h_new = tanh(emb[tok]@Wih^T + h@Whh^T + b_d) -- 256 tiles
    if (gw < 256) {
      const int mTile = gw >> 5, nTile = gw & 31;
      const int n = nTile * 16 + laneN;
      v8f acc;
      const float bb = b_d[n];
#pragma unroll
      for (int r = 0; r < 8; ++r) acc[r] = bb;

      const int b_row = mTile * 16 + laneN;
      const bf16* erow = embbf + (size_t)tok[b_row] * Hn;  // per-lane gathered row
#pragma unroll 4
      for (int kk = 0; kk < Hn / 32; ++kk) {
        const int kb = kk * 32 + half * 8;
        union { v16bf v; v8bf h[2]; } u;
        u.h[0] = *(const v8bf*)(erow + kb);
        u.h[1] = *(const v8bf*)(erow + kb + 16);
        v16bf bfr = load_bfrag(pWih_d, Hn / 32, nTile, kk, lane);
        acc = WMMA_BF16(u.v, bfr, acc);
      }
      const bf16* Abase = hprev + (size_t)mTile * 16 * Hn;
#pragma unroll 4
      for (int kk = 0; kk < Hn / 32; ++kk) {
        v16bf a   = load_afrag(Abase, Hn, kk, lane);
        v16bf bfr = load_bfrag(pWhh_d, Hn / 32, nTile, kk, lane);
        acc = WMMA_BF16(a, bfr, acc);
      }
#pragma unroll
      for (int r = 0; r < 8; ++r) {
        int b_r = mTile * 16 + r + 8 * half;
        float hv = tanhf(acc[r]);
        bf16 hb = f2bf(hv);
        hnext[(size_t)b_r * Hn + n] = hb;
        hc[(size_t)b_r * (2 * Hn) + n] = hb;  // left half of [h, ctx]
        if (st == Tn - 1) out_hT[(size_t)b_r * Hn + n] = hv;
      }
    }
    grid_barrier(cnt, gen, DEC_WG);

    // -- phase B: q = h_new@Wq^T + bq (fp32) -- 256 tiles
    if (gw < 256) {
      const int mTile = gw >> 5, nTile = gw & 31;
      const int n = nTile * 16 + laneN;
      v8f acc;
      const float bb = bq[n];
#pragma unroll
      for (int r = 0; r < 8; ++r) acc[r] = bb;
      const bf16* Abase = hnext + (size_t)mTile * 16 * Hn;
#pragma unroll 4
      for (int kk = 0; kk < Hn / 32; ++kk) {
        v16bf a   = load_afrag(Abase, Hn, kk, lane);
        v16bf bfr = load_bfrag(pWq, Hn / 32, nTile, kk, lane);
        acc = WMMA_BF16(a, bfr, acc);
      }
#pragma unroll
      for (int r = 0; r < 8; ++r) {
        int b_r = mTile * 16 + r + 8 * half;
        qbuf[(size_t)b_r * Hn + n] = acc[r];
      }
    }
    grid_barrier(cnt, gen, DEC_WG);

    // -- phase C: scores[b][s] = values[b][s][:] . q[b][:] (L2-resident sweep)
    for (int i = tid; i < Bn * Sn; i += nthr) {
      int b = i >> 8, s = i & (Sn - 1);
      const bf16*  vrow = values + ((size_t)b * Sn + s) * Hn;
      const float* qv   = qbuf + (size_t)b * Hn;
      float acc = 0.f;
#pragma unroll 8
      for (int h0 = 0; h0 < Hn; h0 += 8) {
        v8bf vv = *(const v8bf*)(vrow + h0);
#pragma unroll
        for (int j = 0; j < 8; ++j) acc += bf2f(vv[j]) * qv[h0 + j];
      }
      scores[i] = acc;
    }
    grid_barrier(cnt, gen, DEC_WG);

    // -- phase D: softmax over S, one wave per batch row
    if (gw < Bn) {
      const int b = gw;
      float v[Sn / 32];
      float m = -3.4e38f;
#pragma unroll
      for (int j = 0; j < Sn / 32; ++j) {
        v[j] = scores[(size_t)b * Sn + lane + 32 * j];
        m = fmaxf(m, v[j]);
      }
      m = wave_max(m);
      float sum = 0.f;
#pragma unroll
      for (int j = 0; j < Sn / 32; ++j) { v[j] = __expf(v[j] - m); sum += v[j]; }
      sum = wave_sum(sum);
      const float inv = 1.f / sum;
#pragma unroll
      for (int j = 0; j < Sn / 32; ++j) {
        int s = lane + 32 * j;
        float wv = v[j] * inv;
        wbuf[(size_t)b * Sn + s] = wv;
        out_att[((size_t)b * Tn + st) * Sn + s] = wv;
      }
    }
    grid_barrier(cnt, gen, DEC_WG);

    // -- phase E: ctx[b][h] = sum_s w[b][s] * values[b][s][h]
    for (int i = tid; i < Bn * Hn; i += nthr) {
      int b = i >> 9, hh = i & (Hn - 1);
      const float* wr = wbuf + (size_t)b * Sn;
      const bf16*  vb = values + (size_t)b * Sn * Hn + hh;
      float acc = 0.f;
#pragma unroll 4
      for (int s = 0; s < Sn; ++s) acc += wr[s] * bf2f(vb[(size_t)s * Hn]);
      hc[(size_t)b * (2 * Hn) + Hn + hh] = f2bf(acc);  // right half of [h, ctx]
    }
    grid_barrier(cnt, gen, DEC_WG);

    // -- phase F: logits = relu([h,ctx]@Wfc^T + bfc) -- 256 tiles, K=1024
    if (gw < 256) {
      const int mTile = gw >> 5, nTile = gw & 31;
      const int n = nTile * 16 + laneN;
      v8f acc;
      const float bb = bfc[n];
#pragma unroll
      for (int r = 0; r < 8; ++r) acc[r] = bb;
      const bf16* Abase = hc + (size_t)mTile * 16 * (2 * Hn);
#pragma unroll 4
      for (int kk = 0; kk < (2 * Hn) / 32; ++kk) {
        v16bf a   = load_afrag(Abase, 2 * Hn, kk, lane);
        v16bf bfr = load_bfrag(pWfc, (2 * Hn) / 32, nTile, kk, lane);
        acc = WMMA_BF16(a, bfr, acc);
      }
#pragma unroll
      for (int r = 0; r < 8; ++r) {
        int b_r = mTile * 16 + r + 8 * half;
        out_vec[((size_t)b_r * Tn + st) * Vn + n] = fmaxf(acc[r], 0.f);
      }
    }
    grid_barrier(cnt, gen, DEC_WG);

    // -- phase G: tok[b] = argmax_v logits (first-occurrence ties), wave/b
    if (gw < Bn) {
      const int b = gw;
      const float* lr = out_vec + ((size_t)b * Tn + st) * Vn;
      float bv = -1.f;
      int bi = 0;
#pragma unroll
      for (int j = 0; j < Vn / 32; ++j) {
        int idx = lane + 32 * j;
        float vv = lr[idx];
        if (vv > bv) { bv = vv; bi = idx; }
      }
      for (int m = 16; m > 0; m >>= 1) {
        float ov = __shfl_xor(bv, m, 32);
        int   oi = __shfl_xor(bi, m, 32);
        if (ov > bv || (ov == bv && oi < bi)) { bv = ov; bi = oi; }
      }
      if (lane == 0) tok[b] = bi;
    }
    grid_barrier(cnt, gen, DEC_WG);

    bf16* tmp = hprev; hprev = hnext; hnext = tmp;
  }
}

// ---------------------------------------------------------------------------
extern "C" void kernel_launch(void* const* d_in, const int* in_sizes, int n_in,
                              void* d_out, int out_size, void* d_ws, size_t ws_size,
                              hipStream_t stream) {
  const float* x      = (const float*)d_in[0];
  const float* W_ih_f = (const float*)d_in[1];
  const float* W_hh_f = (const float*)d_in[2];
  const float* b_f    = (const float*)d_in[3];
  const float* W_ih_b = (const float*)d_in[4];
  const float* W_hh_b = (const float*)d_in[5];
  const float* b_b    = (const float*)d_in[6];
  const float* emb    = (const float*)d_in[7];
  const float* W_ih_d = (const float*)d_in[8];
  const float* W_hh_d = (const float*)d_in[9];
  const float* b_d    = (const float*)d_in[10];
  const float* Wq     = (const float*)d_in[11];
  const float* bq     = (const float*)d_in[12];
  const float* We2d   = (const float*)d_in[13];
  const float* be2d   = (const float*)d_in[14];
  const float* Wfc    = (const float*)d_in[15];
  const float* bfc    = (const float*)d_in[16];
  (void)in_sizes; (void)n_in; (void)out_size; (void)ws_size;

  // workspace layout (bytes, 256-aligned pieces)
  char* ws = (char*)d_ws;
  constexpr size_t SZ_PK512  = (size_t)512 * 512 * 2;    // 512KB packed bf16
  constexpr size_t SZ_PK1024 = (size_t)512 * 1024 * 2;   // 1MB
  size_t o = 0;
  auto take = [&](size_t bytes) { size_t r = o; o = (o + bytes + 255) & ~(size_t)255; return r; };
  bf16* pWhh_f = (bf16*)(ws + take(SZ_PK512));
  bf16* pWhh_b = (bf16*)(ws + take(SZ_PK512));
  bf16* pWhh_d = (bf16*)(ws + take(SZ_PK512));
  bf16* pWih_d = (bf16*)(ws + take(SZ_PK512));
  bf16* pWq    = (bf16*)(ws + take(SZ_PK512));
  bf16* pWe2d  = (bf16*)(ws + take(SZ_PK1024));
  bf16* pWfc   = (bf16*)(ws + take(SZ_PK1024));
  bf16* embbf  = (bf16*)(ws + take((size_t)Vn * Hn * 2));
  bf16* valF   = (bf16*)(ws + take((size_t)Bn * Sn * Hn * 2));  // 32MB (becomes `values`)
  bf16* valB   = (bf16*)(ws + take((size_t)Bn * Sn * Hn * 2));  // 32MB
  bf16* hfb    = (bf16*)(ws + take((size_t)Bn * 2 * Hn * 2));
  bf16* hprev  = (bf16*)(ws + take((size_t)Bn * Hn * 2));
  bf16* hnext  = (bf16*)(ws + take((size_t)Bn * Hn * 2));
  bf16* hc     = (bf16*)(ws + take((size_t)Bn * 2 * Hn * 2));
  float* qbuf  = (float*)(ws + take((size_t)Bn * Hn * 4));
  float* scor  = (float*)(ws + take((size_t)Bn * Sn * 4));
  float* wbuf  = (float*)(ws + take((size_t)Bn * Sn * 4));
  int*   tok   = (int*)  (ws + take((size_t)Bn * 4));
  unsigned* bar = (unsigned*)(ws + take(2 * 4));

  float* out_vec = (float*)d_out;                              // (B,T,V)
  float* out_hT  = out_vec + (size_t)Bn * Tn * Vn;             // (1,B,H)
  float* out_att = out_hT + (size_t)Bn * Hn;                   // (B,T,S)

  // 1) pack weights into WMMA B-fragment order (bf16)
  pack_b_kernel<<<512, 256, 0, stream>>>(W_hh_f, pWhh_f, 512, 512);
  pack_b_kernel<<<512, 256, 0, stream>>>(W_hh_b, pWhh_b, 512, 512);
  pack_b_kernel<<<512, 256, 0, stream>>>(W_hh_d, pWhh_d, 512, 512);
  pack_b_kernel<<<512, 256, 0, stream>>>(W_ih_d, pWih_d, 512, 512);
  pack_b_kernel<<<512, 256, 0, stream>>>(Wq,     pWq,    512, 512);
  pack_b_kernel<<<1024, 256, 0, stream>>>(We2d,  pWe2d,  512, 1024);
  pack_b_kernel<<<1024, 256, 0, stream>>>(Wfc,   pWfc,   512, 1024);
  f2bf_kernel<<<512, 256, 0, stream>>>(emb, embbf, Vn * Hn);

  // 2) encoder: both directions in parallel, 256KB h ping-pong + 2KB x stage LDS
  enc_rnn_kernel<<<2, 1024,
                   (size_t)2 * Bn * Hn * sizeof(bf16) + (size_t)2 * 2 * Bn * sizeof(float),
                   stream>>>(
      x, W_ih_f, b_f, W_ih_b, b_b, pWhh_f, pWhh_b, valF, valB, hfb);

  // 3) values = of + ob_rev (in place into valF)
  addval_kernel<<<4096, 256, 0, stream>>>(valF, valB, Bn * Sn * Hn);

  // 4) decoder initial hidden + token/barrier init
  dec_init_kernel<<<32, 256, 0, stream>>>(hfb, pWe2d, be2d, hprev, tok, bar);

  // 5) persistent decoder loop (T=64 steps, grid-barriered phases)
  dec_loop_kernel<<<DEC_WG, DEC_THR, 0, stream>>>(
      embbf, pWih_d, pWhh_d, b_d, pWq, bq, pWfc, bfc,
      valF, hprev, hnext, hc, qbuf, scor, wbuf, tok,
      out_vec, out_hT, out_att, bar);
}
